// RNNModel_46445776339286
// MI455X (gfx1250) — matrix operand order repaired and checked by reading, not compiled, not run on hardware
//
#include <hip/hip_runtime.h>

typedef __attribute__((ext_vector_type(16))) _Float16 v16h;
typedef __attribute__((ext_vector_type(8)))  _Float16 v8h;
typedef __attribute__((ext_vector_type(8)))  float    v8f;

#define HIDDEN   128
#define IN_CH    13
#define CPAD     32
#define OUT_DIM  64
#define TT       10
#define HW       2048     // H*W = 32*64
#define NROWS    65536    // B*H*W
#define MBLK     64       // rows per block
#define NTHREADS 128      // 4 wave32, 16 rows per wave

struct __align__(32) Smem {
    _Float16 wih0[HIDDEN * CPAD];     //  8 KB, W_ih0 padded K 13->32
    _Float16 whh0[HIDDEN * HIDDEN];   // 32 KB
    _Float16 wih1[HIDDEN * HIDDEN];   // 32 KB
    _Float16 whh1[HIDDEN * HIDDEN];   // 32 KB
    _Float16 wfc [HIDDEN * OUT_DIM];  // 16 KB
    _Float16 xbuf[MBLK * CPAD];       //  4 KB, current x_t tile (padded)
    _Float16 h0  [MBLK * HIDDEN];     // 16 KB, layer-0 hidden state (f16)
    _Float16 h1  [MBLK * HIDDEN];     // 16 KB, layer-1 hidden state (f16)
    float    bias0[HIDDEN];           // b_ih0 + b_hh0
    float    bias1[HIDDEN];           // b_ih1 + b_hh1
    float    biasf[OUT_DIM];
};

// A-fragment (16x32 f16, M x K), contiguous K order matching B:
// lane<16 : row M=lane,    K = kb+{0..15}
// lane>=16: row M=lane-16, K = kb+16+{0..15}
__device__ inline v16h load_a_frag(const _Float16* base, int m, int ld, int kb, int half) {
    const _Float16* p = base + m * ld + kb + half * 16;
    v8h lo = *(const v8h*)(p);
    v8h hi = *(const v8h*)(p + 8);
    v16h a;
#pragma unroll
    for (int i = 0; i < 8; ++i) { a[i] = lo[i]; a[i + 8] = hi[i]; }
    return a;
}

// B-fragment (32x16 f16, K x N): lane holds column N = lane%16,
// lanes 0-15 carry K = kb+0..15, lanes 16-31 carry K = kb+16..31.
// Weight rows W[j][k] are K-contiguous, so this is one 32B LDS read.
__device__ inline v16h load_b_frag(const _Float16* base, int j, int ld, int kb, int half) {
    return *(const v16h*)(base + j * ld + kb + half * 16);
}

__device__ inline v8f wmma_f16(v16h a, v16h b, v8f c) {
    return __builtin_amdgcn_wmma_f32_16x16x32_f16(false, a, false, b, (short)0, c,
                                                  false, false);
}

__device__ inline v8f splat8(float v) {
    v8f x;
#pragma unroll
    for (int i = 0; i < 8; ++i) x[i] = v;
    return x;
}

// One K=32 chunk of a 16x(16*NJT) GEMM: preload all NJT B-fragments, then
// issue NJT back-to-back WMMAs (single s_wait covers the batch; XDL pipelining).
template<int NJT>
__device__ inline void gemm_chunk(const _Float16* Abase, int m, int ald, int kb,
                                  const _Float16* Bbase, int bld, int r, int half,
                                  v8f* acc) {
    v16h a = load_a_frag(Abase, m, ald, kb, half);
    v16h b[NJT];
#pragma unroll
    for (int jt = 0; jt < NJT; ++jt)
        b[jt] = load_b_frag(Bbase, jt * 16 + r, bld, kb, half);
#pragma unroll
    for (int jt = 0; jt < NJT; ++jt)
        acc[jt] = wmma_f16(a, b[jt], acc[jt]);
}

// ReLU + f32->f16 store of a wave's 16x128 result tile into LDS state.
__device__ inline void store_relu(v8f* acc, _Float16* dst, int waveRow,
                                  int r, int half) {
#pragma unroll
    for (int jt = 0; jt < 8; ++jt) {
#pragma unroll
        for (int i = 0; i < 8; ++i) {
            int rowl = waveRow + i + half * 8;      // C/D layout row
            float v = acc[jt][i];
            v = v > 0.f ? v : 0.f;
            dst[rowl * HIDDEN + jt * 16 + r] = (_Float16)v;
        }
    }
}

__global__ __launch_bounds__(NTHREADS) void rnn2_wmma_kernel(
    const float* __restrict__ xg,
    const float* __restrict__ Wih0, const float* __restrict__ Whh0,
    const float* __restrict__ bih0, const float* __restrict__ bhh0,
    const float* __restrict__ Wih1, const float* __restrict__ Whh1,
    const float* __restrict__ bih1, const float* __restrict__ bhh1,
    const float* __restrict__ Wfc,  const float* __restrict__ bfc,
    float* __restrict__ out)
{
    __shared__ Smem s;

    const int tid     = threadIdx.x;
    const int wave    = tid >> 5;
    const int lane    = tid & 31;
    const int half    = lane >> 4;      // 0 or 1
    const int r       = lane & 15;      // 0..15
    const int blkRow  = blockIdx.x * MBLK;
    const int waveRow = wave * 16;      // local base row of this wave
    const int m       = waveRow + r;    // A-fragment row for this lane

    // batch index / spatial remainder are block-constant (MBLK divides HW)
    const int bidx = blkRow / HW;
    const int rem0 = blkRow % HW;

    // ---------------- stage weights (f32 -> f16) ----------------
    for (int i = tid; i < HIDDEN * CPAD; i += NTHREADS) {
        int j = i >> 5, c = i & 31;
        s.wih0[i] = (c < IN_CH) ? (_Float16)Wih0[j * IN_CH + c] : (_Float16)0.f;
    }
    for (int i = tid; i < HIDDEN * HIDDEN; i += NTHREADS) {
        s.whh0[i] = (_Float16)Whh0[i];
        s.wih1[i] = (_Float16)Wih1[i];
        s.whh1[i] = (_Float16)Whh1[i];
    }
    for (int i = tid; i < OUT_DIM * HIDDEN; i += NTHREADS)
        s.wfc[i] = (_Float16)Wfc[i];
    for (int i = tid; i < HIDDEN; i += NTHREADS) {
        s.bias0[i] = bih0[i] + bhh0[i];
        s.bias1[i] = bih1[i] + bhh1[i];
    }
    for (int i = tid; i < OUT_DIM; i += NTHREADS)
        s.biasf[i] = bfc[i];
    __syncthreads();
    // From here on each wave only touches its own 16 rows of xbuf/h0/h1:
    // no further barriers needed (DScnt keeps same-wave LDS ops ordered).

    v8f acc[8];

    // =========== t = 0 (h_prev == 0: no recurrent GEMMs) ===========
    for (int i = lane; i < 16 * CPAD; i += 32) {
        int rl = i >> 5, c = i & 31;
        int rowl = waveRow + rl;
        float v = 0.f;
        if (c < IN_CH) {
            v = xg[(size_t)(bidx * TT * IN_CH + c) * HW + rem0 + rowl];
            __builtin_prefetch(                       // warm L2 for t=1
                &xg[(size_t)((bidx * TT + 1) * IN_CH + c) * HW + rem0 + rowl], 0, 0);
        }
        s.xbuf[rowl * CPAD + c] = (_Float16)v;
    }
#pragma unroll
    for (int jt = 0; jt < 8; ++jt) acc[jt] = splat8(s.bias0[jt * 16 + r]);
    gemm_chunk<8>(s.xbuf, m, CPAD, 0, s.wih0, CPAD, r, half, acc);
    store_relu(acc, s.h0, waveRow, r, half);

#pragma unroll
    for (int jt = 0; jt < 8; ++jt) acc[jt] = splat8(s.bias1[jt * 16 + r]);
#pragma unroll
    for (int kb = 0; kb < HIDDEN; kb += 32)
        gemm_chunk<8>(s.h0, m, HIDDEN, kb, s.wih1, HIDDEN, r, half, acc);
    store_relu(acc, s.h1, waveRow, r, half);

    // =========== steady state: t = 1 .. T-1 (branch-free body) ===========
    for (int t = 1; t < TT; ++t) {
        // stage this wave's 16 rows of x_t (zero-pad c to 32)
        for (int i = lane; i < 16 * CPAD; i += 32) {
            int rl = i >> 5, c = i & 31;
            int rowl = waveRow + rl;
            float v = 0.f;
            if (c < IN_CH) {
                v = xg[(size_t)((bidx * TT + t) * IN_CH + c) * HW + rem0 + rowl];
                if (t + 1 < TT)
                    __builtin_prefetch(
                        &xg[(size_t)((bidx * TT + t + 1) * IN_CH + c) * HW + rem0 + rowl],
                        0, 0);
            }
            s.xbuf[rowl * CPAD + c] = (_Float16)v;
        }

        // ---- layer 0: relu(x_t @ Wih0^T + h0 @ Whh0^T + bias0) ----
#pragma unroll
        for (int jt = 0; jt < 8; ++jt) acc[jt] = splat8(s.bias0[jt * 16 + r]);
        gemm_chunk<8>(s.xbuf, m, CPAD, 0, s.wih0, CPAD, r, half, acc);
#pragma unroll
        for (int kb = 0; kb < HIDDEN; kb += 32)
            gemm_chunk<8>(s.h0, m, HIDDEN, kb, s.whh0, HIDDEN, r, half, acc);
        store_relu(acc, s.h0, waveRow, r, half);

        // ---- layer 1: relu(h0 @ Wih1^T + h1 @ Whh1^T + bias1) ----
#pragma unroll
        for (int jt = 0; jt < 8; ++jt) acc[jt] = splat8(s.bias1[jt * 16 + r]);
#pragma unroll
        for (int kb = 0; kb < HIDDEN; kb += 32)
            gemm_chunk<8>(s.h0, m, HIDDEN, kb, s.wih1, HIDDEN, r, half, acc);
#pragma unroll
        for (int kb = 0; kb < HIDDEN; kb += 32)
            gemm_chunk<8>(s.h1, m, HIDDEN, kb, s.whh1, HIDDEN, r, half, acc);
        store_relu(acc, s.h1, waveRow, r, half);
    }

    // ---------------- final FC: h1_T @ W_fc^T + b_fc ----------------
    v8f accf[4];
#pragma unroll
    for (int jt = 0; jt < 4; ++jt) accf[jt] = splat8(s.biasf[jt * 16 + r]);
#pragma unroll
    for (int kb = 0; kb < HIDDEN; kb += 32)
        gemm_chunk<4>(s.h1, m, HIDDEN, kb, s.wfc, HIDDEN, r, half, accf);
#pragma unroll
    for (int jt = 0; jt < 4; ++jt) {
#pragma unroll
        for (int i = 0; i < 8; ++i) {
            int rowg = blkRow + waveRow + i + half * 8;
            out[(size_t)rowg * OUT_DIM + jt * 16 + r] = accf[jt][i];
        }
    }
}

extern "C" void kernel_launch(void* const* d_in, const int* in_sizes, int n_in,
                              void* d_out, int out_size, void* d_ws, size_t ws_size,
                              hipStream_t stream) {
    (void)in_sizes; (void)n_in; (void)out_size; (void)d_ws; (void)ws_size;
    const float* xg   = (const float*)d_in[0];
    const float* Wih0 = (const float*)d_in[1];
    const float* Whh0 = (const float*)d_in[2];
    const float* bih0 = (const float*)d_in[3];
    const float* bhh0 = (const float*)d_in[4];
    const float* Wih1 = (const float*)d_in[5];
    const float* Whh1 = (const float*)d_in[6];
    const float* bih1 = (const float*)d_in[7];
    const float* bhh1 = (const float*)d_in[8];
    const float* Wfc  = (const float*)d_in[9];
    const float* bfc  = (const float*)d_in[10];
    float* out = (float*)d_out;

    dim3 grid(NROWS / MBLK);   // 1024 blocks
    dim3 block(NTHREADS);      // 4 wave32 each
    rnn2_wmma_kernel<<<grid, block, 0, stream>>>(
        xg, Wih0, Whh0, bih0, bhh0, Wih1, Whh1, bih1, bhh1, Wfc, bfc, out);
}
